// Attn_59940563583594
// MI455X (gfx1250) — matrix operand order repaired and checked by reading, not compiled, not run on hardware
//
#include <hip/hip_runtime.h>
#include <hip/hip_bf16.h>

// ---------------------------------------------------------------------------
// MI455X (gfx1250) fused attention block, bf16 WMMA + Tensor Data Mover.
//   q,k,v = hidden @ W{q,k,v}.T + b   (TDM-staged A tile, 16x96 per wave)
//   w     = softmax(mask(q @ k.T))    (WMMA scores -> LDS -> wave32 softmax)
//   h     = w @ v                      (TDM-staged P tile, causal K trunc)
//   out   = relu(h @ W1.T + b1) @ W2.T + b2
// ---------------------------------------------------------------------------

#define NB 32
#define NS 1024
#define NE 768
#define NH 64
#define NL 2
#define NM (NB * NS)  // 32768 rows

typedef __attribute__((ext_vector_type(16))) __bf16 v16bf;
typedef __attribute__((ext_vector_type(8)))  float  v8f;
typedef __attribute__((ext_vector_type(4)))  unsigned int v4u;
typedef __attribute__((ext_vector_type(8)))  unsigned int v8u;

union BF8 { uint4 u; __bf16 b[8]; };

__device__ __forceinline__ unsigned short f32_to_bf16(float f) {
  unsigned int u = __float_as_uint(f);
  unsigned int r = (u + 0x7FFFu + ((u >> 16) & 1u)) >> 16;  // round-nearest-even
  return (unsigned short)r;
}
__device__ __forceinline__ unsigned int pack2(float a, float b) {
  return (unsigned int)f32_to_bf16(a) | ((unsigned int)f32_to_bf16(b) << 16);
}

// ---------------------------------------------------------------------------
// Tensor Data Mover: 2-D tile load global -> LDS (data_size = 2 bytes).
// D# per CDNA5 ISA.  Issued by one wave; waiter
// uses s_wait_tensorcnt then the block barriers.
// ---------------------------------------------------------------------------
__device__ __forceinline__ void tdm_load_2d_bf16(unsigned lds_off, const void* gaddr,
                                                 unsigned tile_d0, unsigned tile_d1,
                                                 unsigned stride_d0) {
  unsigned long long ga = (unsigned long long)(size_t)gaddr;
  v4u g0;
  g0[0] = 1u;                                    // count=1, user descriptor
  g0[1] = lds_off;                               // lds_addr [63:32]
  g0[2] = (unsigned)(ga & 0xFFFFFFFFu);          // global_addr[31:0]
  g0[3] = (unsigned)((ga >> 32) & 0x01FFFFFFu)   // global_addr[56:32]
          | (2u << 30);                          // type=2 ("image")
  v8u g1;
  g1[0] = 1u << 16;                              // wg_mask=0, data_size=1 (2B)
  g1[1] = (tile_d0 & 0xFFFFu) << 16;             // tensor_dim0[15:0]
  g1[2] = ((tile_d0 >> 16) & 0xFFFFu)            // tensor_dim0[31:16]
          | ((tile_d1 & 0xFFFFu) << 16);         // tensor_dim1[15:0]
  g1[3] = ((tile_d1 >> 16) & 0xFFFFu)            // tensor_dim1[31:16]
          | (tile_d0 << 16);                     // tile_dim0
  g1[4] = tile_d1 & 0xFFFFu;                     // tile_dim1 (tile_dim2=0)
  g1[5] = stride_d0;                             // tensor_dim0_stride[31:0]
  g1[6] = 0u;                                    // stride hi, dim1_stride lo
  g1[7] = 0u;
  v4u g2 = {0u, 0u, 0u, 0u};                     // dims 2..4 unused (zero)
  v4u g3 = {0u, 0u, 0u, 0u};
  asm volatile("tensor_load_to_lds %0, %1, %2, %3"
               :: "s"(g0), "s"(g1), "s"(g2), "s"(g3) : "memory");
}

__device__ __forceinline__ unsigned lds_offset_of(const void* p) {
  return (unsigned)(size_t)p;  // generic LDS addr: low 32 bits = LDS offset
}

// A fragment: 16x32 (MxK) bf16, row-major source, stride ld (elements).
// lane L: m = L&15, half = L>>4; elems {K=kb+8h+0..7} then {K=kb+16+8h+0..7}.
__device__ __forceinline__ v16bf load_a_frag(const unsigned short* base, int row0,
                                             int ld, int kb) {
  const int lane = threadIdx.x & 31;
  const int m = lane & 15, half = lane >> 4;
  const unsigned short* p = base + (size_t)(row0 + m) * ld + kb + 8 * half;
  BF8 lo, hi;
  lo.u = *(const uint4*)p;
  hi.u = *(const uint4*)(p + 16);
  v16bf a;
#pragma unroll
  for (int i = 0; i < 8; i++) { a[i] = lo.b[i]; a[i + 8] = hi.b[i]; }
  return a;
}

// B fragment: 32x16 (KxN) bf16. Source stored as N x K row-major ("rows are
// columns of B"), stride ld. lane L: n = L&15, half = L>>4; elems {K=kb+8h+0..7} then {K=kb+16+8h+0..7}.
__device__ __forceinline__ v16bf load_b_frag(const unsigned short* baseT, int col0,
                                             int ld, int kb) {
  const int lane = threadIdx.x & 31;
  const int n = lane & 15, half = lane >> 4;
  const unsigned short* p = baseT + (size_t)(col0 + n) * ld + kb + 8 * half;
  BF8 lo, hi;
  lo.u = *(const uint4*)p;
  hi.u = *(const uint4*)(p + 16);
  v16bf b;
#pragma unroll
  for (int i = 0; i < 8; i++) { b[i] = lo.b[i]; b[i + 8] = hi.b[i]; }
  return b;
}

__device__ __forceinline__ v8f wmma_bf16(v16bf a, v16bf b, v8f c) {
  return __builtin_amdgcn_wmma_f32_16x16x32_bf16(false, a, false, b, (short)0, c,
                                                 false, false);
}

// ---------------------------------------------------------------------------
// fp32 -> bf16 convert (4 elements/thread)
// ---------------------------------------------------------------------------
__global__ void __launch_bounds__(256) cvt_bf16_kernel(const float* __restrict__ src,
                                                       unsigned short* __restrict__ dst,
                                                       int n) {
  int i = (blockIdx.x * 256 + threadIdx.x) * 4;
  if (i < n) {
    float4 f = *(const float4*)(src + i);
    unsigned int lo = pack2(f.x, f.y);
    unsigned int hi = pack2(f.z, f.w);
    *(uint2*)(dst + i) = make_uint2(lo, hi);
  }
}

// ---------------------------------------------------------------------------
// QKV projections: X[32768,768] @ W.T + b.
// grid (2048, 1, 3), block 256 (8 waves).  A tile (16x768, 24KB) staged in
// LDS by the Tensor Data Mover; each wave computes a 16x96 strip (6 accums).
// V is stored transposed per batch: Vt[b][e][s].
// ---------------------------------------------------------------------------
__global__ void __launch_bounds__(256)
qkv_gemm_kernel(const unsigned short* __restrict__ Xb,
                const unsigned short* __restrict__ WqB,
                const unsigned short* __restrict__ WkB,
                const unsigned short* __restrict__ WvB,
                const float* __restrict__ bq, const float* __restrict__ bk,
                const float* __restrict__ bv,
                unsigned short* __restrict__ Qo, unsigned short* __restrict__ Ko,
                unsigned short* __restrict__ Vto) {
  __shared__ __align__(16) unsigned short xtile[16 * NE];  // 24 KB
  const int mt = blockIdx.x;                       // 0..2047
  const int wave = threadIdx.x >> 5, lane = threadIdx.x & 31;
  const int which = blockIdx.z;
  const unsigned short* W = (which == 0) ? WqB : (which == 1) ? WkB : WvB;
  const float* bias = (which == 0) ? bq : (which == 1) ? bk : bv;

  if (wave == 0) {
    tdm_load_2d_bf16(lds_offset_of(xtile), Xb + (size_t)mt * 16 * NE, NE, 16, NE);
    __builtin_amdgcn_s_wait_tensorcnt(0);
  }
  __syncthreads();

  const int nt0 = wave * 6;                        // 8 waves x 6 = 48 N-tiles
  v8f c[6] = {};
#pragma unroll 2
  for (int kb = 0; kb < NE; kb += 32) {
    v16bf a = load_a_frag(xtile, 0, NE, kb);       // LDS (ds_load_b128)
#pragma unroll
    for (int t = 0; t < 6; t++) {
      v16bf b = load_b_frag(W, (nt0 + t) * 16, NE, kb);
      c[t] = wmma_bf16(a, b, c[t]);
    }
  }
  const int n = lane & 15, half = lane >> 4;
#pragma unroll
  for (int t = 0; t < 6; t++) {
    const float bval = bias[(nt0 + t) * 16 + n];
    if (which == 2) {
      // transposed store: Vt[b][e][s], 8 consecutive s values -> one b128.
      const int bb = mt / (NS / 16);
      const int s0 = (mt % (NS / 16)) * 16 + 8 * half;
      unsigned short* dst =
          Vto + ((size_t)bb * NE + (nt0 + t) * 16 + n) * NS + s0;
      uint4 u;
      u.x = pack2(c[t][0] + bval, c[t][1] + bval);
      u.y = pack2(c[t][2] + bval, c[t][3] + bval);
      u.z = pack2(c[t][4] + bval, c[t][5] + bval);
      u.w = pack2(c[t][6] + bval, c[t][7] + bval);
      *(uint4*)dst = u;
    } else {
      unsigned short* O = (which == 0) ? Qo : Ko;
#pragma unroll
      for (int r = 0; r < 8; r++)
        O[(size_t)(mt * 16 + r + 8 * half) * NE + (nt0 + t) * 16 + n] =
            f32_to_bf16(c[t][r] + bval);
    }
  }
}

// ---------------------------------------------------------------------------
// scores + causal mask + attention-mask bias + softmax -> P (bf16).
// grid (64, 32) = (q-tile, batch), block 256 = 8 waves.
// Waves cover 4 key tiles each per pass (A fragment reused 4x), causal tiles
// skipped; then wave w normalizes rows 2w/2w+1 with 16-lane reductions.
// ---------------------------------------------------------------------------
__global__ void __launch_bounds__(256)
score_softmax_kernel(const unsigned short* __restrict__ Q,
                     const unsigned short* __restrict__ K,
                     const float* __restrict__ amask,
                     unsigned short* __restrict__ P) {
  __shared__ float sc[16][NS];  // 64 KB (WGP has 320 KB LDS)
  const int qt = blockIdx.x;    // query tile 0..63
  const int b = blockIdx.y;     // batch
  const int wave = threadIdx.x >> 5, lane = threadIdx.x & 31;
  const int n = lane & 15, half = lane >> 4;
  const size_t rowbase = (size_t)b * NS;
  const unsigned short* Qb = Q + rowbase * NE;
  const unsigned short* Kb = K + rowbase * NE;
  const int q0 = qt * 16;

  for (int jb = 0; jb <= qt; jb += 32) {          // 8 waves x 4 tiles per pass
    const int j0 = jb + wave * 4;
    if (j0 <= qt) {                                // wave-uniform guard
      v8f c[4] = {};
#pragma unroll 2
      for (int kb = 0; kb < NE; kb += 32) {
        v16bf a = load_a_frag(Qb, q0, NE, kb);
#pragma unroll
        for (int t = 0; t < 4; t++) {
          if (j0 + t <= qt) {                      // wave-uniform
            v16bf bb = load_b_frag(Kb, (j0 + t) * 16, NE, kb);
            c[t] = wmma_bf16(a, bb, c[t]);
          }
        }
      }
#pragma unroll
      for (int t = 0; t < 4; t++)
        if (j0 + t <= qt)
#pragma unroll
          for (int r = 0; r < 8; r++)
            sc[r + 8 * half][(j0 + t) * 16 + n] = c[t][r];
    }
  }
  __syncthreads();

  // softmax: wave w owns rows 2w (lanes 0-15) and 2w+1 (lanes 16-31);
  // sublane n covers 64 contiguous columns.  wave32 -> width-16 shuffles.
  const int row = 2 * wave + half;
  const int qrow = q0 + row;
  const int c0 = n * 64;
  float mx = -3.0e38f;
  for (int i = 0; i < 64; i++) {
    int col = c0 + i;
    if (col > qrow) break;  // causal
    float x = sc[row][col] + (1.0f - amask[rowbase + col]) * -10000.0f;
    sc[row][col] = x;
    mx = fmaxf(mx, x);
  }
#pragma unroll
  for (int m = 1; m < 16; m <<= 1) mx = fmaxf(mx, __shfl_xor(mx, m, 16));

  float sum = 0.0f;
  for (int i = 0; i < 64; i++) {
    int col = c0 + i;
    float e = 0.0f;
    if (col <= qrow) e = __expf(sc[row][col] - mx);
    sc[row][col] = e;
    sum += e;
  }
#pragma unroll
  for (int m = 1; m < 16; m <<= 1) sum += __shfl_xor(sum, m, 16);
  const float inv = 1.0f / sum;

  unsigned short* dst = P + ((size_t)b * NS + qrow) * NS + c0;
#pragma unroll
  for (int i = 0; i < 64; i += 8) {
    uint4 u;
    u.x = pack2(sc[row][c0 + i + 0] * inv, sc[row][c0 + i + 1] * inv);
    u.y = pack2(sc[row][c0 + i + 2] * inv, sc[row][c0 + i + 3] * inv);
    u.z = pack2(sc[row][c0 + i + 4] * inv, sc[row][c0 + i + 5] * inv);
    u.w = pack2(sc[row][c0 + i + 6] * inv, sc[row][c0 + i + 7] * inv);
    *(uint4*)(dst + i) = u;
  }
}

// ---------------------------------------------------------------------------
// h = P @ V per batch.  grid (64, 1, 32), block 256 (8 waves x 16x96 strip).
// P row tile (16x1024, 32 KB) staged in LDS by the TDM.
// K-loop truncated at causal boundary (P is exactly zero past it).
// ---------------------------------------------------------------------------
__global__ void __launch_bounds__(256)
pv_gemm_kernel(const unsigned short* __restrict__ P,
               const unsigned short* __restrict__ Vt,
               unsigned short* __restrict__ Hout) {
  __shared__ __align__(16) unsigned short ptile[16 * NS];  // 32 KB
  const int b = blockIdx.z;
  const int mt = blockIdx.x;  // 0..63 (within batch)
  const int wave = threadIdx.x >> 5, lane = threadIdx.x & 31;
  const unsigned short* Pb = P + (size_t)b * NS * NS;
  const unsigned short* Vb = Vt + (size_t)b * NE * NS;
  const int kend = (((mt + 1) * 16) + 31) & ~31;  // cover cols <= mt*16+15

  if (wave == 0) {
    tdm_load_2d_bf16(lds_offset_of(ptile), Pb + (size_t)mt * 16 * NS, NS, 16, NS);
    __builtin_amdgcn_s_wait_tensorcnt(0);
  }
  __syncthreads();

  const int nt0 = wave * 6;  // 8 waves x 6 = 48 N-tiles
  v8f c[6] = {};
  for (int kb = 0; kb < kend; kb += 32) {
    v16bf a = load_a_frag(ptile, 0, NS, kb);       // LDS
#pragma unroll
    for (int t = 0; t < 6; t++) {
      v16bf bb = load_b_frag(Vb, (nt0 + t) * 16, NS, kb);  // rows of Vt
      c[t] = wmma_bf16(a, bb, c[t]);
    }
  }
  const int n = lane & 15, half = lane >> 4;
  const size_t grow = (size_t)b * NS + mt * 16;
#pragma unroll
  for (int t = 0; t < 6; t++)
#pragma unroll
    for (int r = 0; r < 8; r++)
      Hout[(grow + r + 8 * half) * NE + (nt0 + t) * 16 + n] =
          f32_to_bf16(c[t][r]);
}

// ---------------------------------------------------------------------------
// h1 = relu(h @ W1.T + b1)  (M=32768, N=64, K=768).  grid 256, block 256;
// wave = one 16x64 strip (4 accumulators, A fragment reused 4x).
// ---------------------------------------------------------------------------
__global__ void __launch_bounds__(256)
mlp1_kernel(const unsigned short* __restrict__ Hb,
            const unsigned short* __restrict__ W1b,
            const float* __restrict__ b1, float* __restrict__ H1) {
  const int wave = threadIdx.x >> 5, lane = threadIdx.x & 31;
  const int mt = blockIdx.x * 8 + wave;  // 0..2047
  v8f c[4] = {};
#pragma unroll 2
  for (int kb = 0; kb < NE; kb += 32) {
    v16bf a = load_a_frag(Hb, mt * 16, NE, kb);
#pragma unroll
    for (int t = 0; t < 4; t++) {
      v16bf bb = load_b_frag(W1b, t * 16, NE, kb);
      c[t] = wmma_bf16(a, bb, c[t]);
    }
  }
  const int n = lane & 15, half = lane >> 4;
#pragma unroll
  for (int t = 0; t < 4; t++) {
    const float bval = b1[t * 16 + n];
#pragma unroll
    for (int r = 0; r < 8; r++) {
      float v = fmaxf(c[t][r] + bval, 0.0f);
      H1[(size_t)(mt * 16 + r + 8 * half) * NH + t * 16 + n] = v;
    }
  }
}

// ---------------------------------------------------------------------------
// out = h1 @ W2.T + b2  (N=2: trivial fp32 dot, memory-bound)
// ---------------------------------------------------------------------------
__global__ void __launch_bounds__(256)
mlp2_kernel(const float* __restrict__ H1, const float* __restrict__ W2,
            const float* __restrict__ b2, float* __restrict__ out) {
  const int m = blockIdx.x * 256 + threadIdx.x;
  if (m >= NM) return;
  const float* h = H1 + (size_t)m * NH;
  float a0 = b2[0], a1 = b2[1];
#pragma unroll 8
  for (int k = 0; k < NH; k++) {
    float hv = h[k];
    a0 += hv * W2[k];
    a1 += hv * W2[NH + k];
  }
  out[(size_t)m * 2 + 0] = a0;
  out[(size_t)m * 2 + 1] = a1;
}

// ---------------------------------------------------------------------------
extern "C" void kernel_launch(void* const* d_in, const int* in_sizes, int n_in,
                              void* d_out, int out_size, void* d_ws, size_t ws_size,
                              hipStream_t stream) {
  const float* hidden = (const float*)d_in[0];
  const float* amask  = (const float*)d_in[1];
  const float* Wk = (const float*)d_in[2];
  const float* bk = (const float*)d_in[3];
  const float* Wq = (const float*)d_in[4];
  const float* bq = (const float*)d_in[5];
  const float* Wv = (const float*)d_in[6];
  const float* bv = (const float*)d_in[7];
  const float* W1 = (const float*)d_in[8];
  const float* b1 = (const float*)d_in[9];
  const float* W2 = (const float*)d_in[10];
  const float* b2 = (const float*)d_in[11];
  float* out = (float*)d_out;

  char* ws = (char*)d_ws;
  size_t off = 0;
  auto alloc = [&](size_t bytes) -> char* {
    char* p = ws + off;
    off = (off + bytes + 255) & ~(size_t)255;
    return p;
  };
  unsigned short* hB  = (unsigned short*)alloc((size_t)NM * NE * 2);
  unsigned short* WqB = (unsigned short*)alloc((size_t)NE * NE * 2);
  unsigned short* WkB = (unsigned short*)alloc((size_t)NE * NE * 2);
  unsigned short* WvB = (unsigned short*)alloc((size_t)NE * NE * 2);
  unsigned short* W1B = (unsigned short*)alloc((size_t)NH * NE * 2);
  unsigned short* Qb  = (unsigned short*)alloc((size_t)NM * NE * 2);
  unsigned short* Kb  = (unsigned short*)alloc((size_t)NM * NE * 2);
  unsigned short* Vtb = (unsigned short*)alloc((size_t)NM * NE * 2);
  unsigned short* Pb  = (unsigned short*)alloc((size_t)NB * NS * NS * 2);
  unsigned short* Hb  = (unsigned short*)alloc((size_t)NM * NE * 2);
  float*          H1  = (float*)alloc((size_t)NM * NH * 4);

  auto cvt = [&](const float* s, unsigned short* d, int n) {
    cvt_bf16_kernel<<<dim3((n / 4 + 255) / 256), dim3(256), 0, stream>>>(s, d, n);
  };
  cvt(hidden, hB, NM * NE);
  cvt(Wq, WqB, NE * NE);
  cvt(Wk, WkB, NE * NE);
  cvt(Wv, WvB, NE * NE);
  cvt(W1, W1B, NH * NE);

  qkv_gemm_kernel<<<dim3(NM / 16, 1, 3), dim3(256), 0, stream>>>(
      hB, WqB, WkB, WvB, bq, bk, bv, Qb, Kb, Vtb);

  score_softmax_kernel<<<dim3(NS / 16, NB), dim3(256), 0, stream>>>(Qb, Kb, amask, Pb);

  pv_gemm_kernel<<<dim3(NS / 16, 1, NB), dim3(256), 0, stream>>>(Pb, Vtb, Hb);

  mlp1_kernel<<<dim3(NM / 16 / 8), dim3(256), 0, stream>>>(Hb, W1B, b1, H1);

  mlp2_kernel<<<dim3((NM + 255) / 256), dim3(256), 0, stream>>>(H1, W2, b2, out);
}
